// Attention_71571335021109
// MI455X (gfx1250) — matrix operand order repaired and checked by reading, not compiled, not run on hardware
//
#include <hip/hip_runtime.h>
#include <hip/hip_bf16.h>

// MI455X / gfx1250 fused attention:
//   cvt kernels -> GEMM1(x * [wq|wk|wv], RoPE epilogue, bf16 out)
//   -> flash attention (online softmax, WMMA bf16)
//   -> GEMM2(o * wo, fp32 out)
// Matmuls: v_wmma_f32_16x16x32_bf16 (wave32, 16x16 tiles, K=32).
// Straight global->LDS tile copies use gfx1250 async loads
// (global_load_async_to_lds_b128, ASYNCcnt) instead of VGPR staging.

typedef unsigned short u16;
typedef unsigned int   u32;
typedef __attribute__((ext_vector_type(8)))  float  v8f;
typedef __attribute__((ext_vector_type(16))) __bf16 v16bf;

union BFrag {
  v16bf bf;
  uint4 u[2];
};

__device__ __forceinline__ u16 f2bf(float f) {
  // round-to-nearest-even f32 -> bf16
  u32 u = __float_as_uint(f);
  u32 r = u + 0x7FFFu + ((u >> 16) & 1u);
  return (u16)(r >> 16);
}

__device__ __forceinline__ v8f wmma_bf16(const BFrag& a, const BFrag& b, v8f c) {
  // (neg_a, A, neg_b, B, c_mod, C, reuse_a, reuse_b)
  return __builtin_amdgcn_wmma_f32_16x16x32_bf16(false, a.bf, false, b.bf,
                                                 (short)0, c, false, false);
}

// gfx1250 async global->LDS copy, 16 bytes per lane. VDST VGPR carries the
// LDS byte address (low 32 bits of the generic shared pointer); tracked with
// ASYNCcnt, no VGPR data round-trip.
__device__ __forceinline__ void async_b128(void* lds, const void* gaddr) {
  u32 laddr = (u32)(size_t)lds;
  unsigned long long ga = (unsigned long long)(size_t)gaddr;
  asm volatile("global_load_async_to_lds_b128 %0, %1, off"
               :
               : "v"(laddr), "v"(ga)
               : "memory");
}

__device__ __forceinline__ void wait_async0() {
  asm volatile("s_wait_asynccnt 0x0" ::: "memory");
}

// ---------------------------------------------------------------- conversions

__global__ void k_f32_to_bf16(const float* __restrict__ s, u16* __restrict__ d,
                              int n) {
  int i = blockIdx.x * 256 + threadIdx.x;
  if (i < n) d[i] = f2bf(s[i]);
}

// wq/wk/wv [1024][1024] -> fused bf16 weight [1024][3072], col = proj*1024 + h*64 + e
__global__ void k_pack_qkv(const float* __restrict__ wq,
                           const float* __restrict__ wk,
                           const float* __restrict__ wv, u16* __restrict__ d) {
  int i = blockIdx.x * 256 + threadIdx.x;   // exact: 3*1024*1024 threads
  int proj = i >> 20;
  int rem  = i & 1048575;
  int dd   = rem >> 10;
  int c    = rem & 1023;
  const float* src = (proj == 0) ? wq : (proj == 1) ? wk : wv;
  d[(size_t)dd * 3072 + proj * 1024 + c] = f2bf(src[rem]);
}

// ------------------------------------------------------------------- GEMM
// C[M x N] = A[M x K(bf16)] * B[K x N(bf16)], block tile 128x64, 256 thr (8 waves).
// Wave w computes rows [w*16, w*16+16) x all 64 cols (4 accumulator tiles).
// MODE 0: epilogue applies RoPE (cols are one (proj,head)), writes bf16 q/k/v.
// MODE 1: writes fp32 out.
template <int MODE>
__global__ __launch_bounds__(256, 1) void gemm_bf16_128x64(
    const u16* __restrict__ A, const u16* __restrict__ Bw, int K, int N,
    u16* __restrict__ qd, u16* __restrict__ kd, u16* __restrict__ vd,
    float* __restrict__ outf) {
  __shared__ __align__(16) u16 lA[128 * 32];  // [row][k], k-contiguous
  __shared__ __align__(16) u16 lB[64 * 32];   // [n][k] (transposed), k-contiguous

  const int tid  = threadIdx.x;
  const int lane = tid & 31;
  const int w    = tid >> 5;
  const int hi   = lane >> 4;   // lane group: rows/K upper half selector
  const int nn   = lane & 15;
  const int M0   = blockIdx.y * 128;
  const int N0   = blockIdx.x * 64;

  v8f zero = {0.f, 0.f, 0.f, 0.f, 0.f, 0.f, 0.f, 0.f};
  v8f acc[4] = {zero, zero, zero, zero};

  const int nk = K >> 5;
  for (int ks = 0; ks < nk; ++ks) {
    const int k0 = ks << 5;
    // A tile 128x32 bf16: straight copy -> async b128, 512 chunks, 2/thread
#pragma unroll
    for (int i = 0; i < 2; ++i) {
      int idx = tid + i * 256;               // 0..511
      int row = idx >> 2, c = (idx & 3) << 3;  // 4 x 8-elem chunks per row
      async_b128(&lA[row * 32 + c],
                 A + (size_t)(M0 + row) * K + k0 + c);
    }
    // B tile 32x64 bf16, transposed into [n][k] (needs VGPR path)
#pragma unroll
    for (int i = 0; i < 8; ++i) {
      int idx = tid + i * 256;
      int kk = idx >> 6, n = idx & 63;
      lB[n * 32 + kk] = Bw[(size_t)(k0 + kk) * N + N0 + n];
    }
    wait_async0();
    __syncthreads();

    BFrag a;  // A 16x32: lane m=nn, VGPR k-pairs contiguous -> 2x ds_load_b128
    const int arow = w * 16 + nn;
    a.u[0] = *(const uint4*)&lA[arow * 32 + hi * 8];
    a.u[1] = *(const uint4*)&lA[arow * 32 + 16 + hi * 8];
#pragma unroll
    for (int nt = 0; nt < 4; ++nt) {
      BFrag b;  // B 32x16: lane n=nn, K pairs contiguous in lB row
      const int nc = nt * 16 + nn;
      b.u[0] = *(const uint4*)&lB[nc * 32 + hi * 8];
      b.u[1] = *(const uint4*)&lB[nc * 32 + 16 + hi * 8];
      acc[nt] = wmma_bf16(a, b, acc[nt]);
    }
    __syncthreads();
  }

  const int rowBase = M0 + w * 16;
  if (MODE == 0) {
    const int proj = N0 >> 10;
    const int h    = (N0 & 1023) >> 6;
    if (proj < 2) {
      // RoPE: pair (e, e+32) == acc tile (t, t+2), same lane, same VGPR.
#pragma unroll
      for (int t = 0; t < 2; ++t) {
#pragma unroll
        for (int r = 0; r < 8; ++r) {
          int   m   = r + hi * 8;
          float pos = (float)((rowBase + m) & 2047);   // s within sequence
          float j   = (float)(t * 16 + nn);            // e half-index [0,32)
          float ang = pos * __expf(-j * 0.28782313665f);  // ln(1e4)/32
          float sn, cs;
          __sincosf(ang, &sn, &cs);
          float f = acc[t][r], sec = acc[t + 2][r];
          acc[t][r]     = f * cs - sec * sn;
          acc[t + 2][r] = sec * cs + f * sn;
        }
      }
    }
    u16* dst = (proj == 0) ? qd : (proj == 1) ? kd : vd;
#pragma unroll
    for (int t = 0; t < 4; ++t)
#pragma unroll
      for (int r = 0; r < 8; ++r) {
        int m = r + hi * 8;
        dst[(size_t)(rowBase + m) * 1024 + h * 64 + t * 16 + nn] =
            f2bf(acc[t][r]);
      }
  } else {
#pragma unroll
    for (int t = 0; t < 4; ++t)
#pragma unroll
      for (int r = 0; r < 8; ++r) {
        int m = r + hi * 8;
        outf[(size_t)(rowBase + m) * N + N0 + t * 16 + nn] = acc[t][r];
      }
  }
}

// --------------------------------------------------------------- flash attn
// grid (S/64, B*H), 128 thr (4 waves). Wave w owns 16 query rows.
// Loops over 2048 keys in 64-wide tiles with online softmax; scores never
// touch HBM. P is converted C-layout -> A-layout through a per-wave LDS
// buffer (same-wave DS ops are in-order, no barrier needed).
// launch_bounds(128,1): lift the VGPR cap so the O/S accumulators + Q
// fragments stay in registers (no scratch spills in the key loop).
__global__ __launch_bounds__(128, 1) void flash_attn(
    const u16* __restrict__ qb, const u16* __restrict__ kb,
    const u16* __restrict__ vb, u16* __restrict__ ob) {
  __shared__ __align__(16) u16 lQ[64 * 64];       // [q][e]
  __shared__ __align__(16) u16 lK[64 * 64];       // [key][e]
  __shared__ __align__(16) u16 lVT[64 * 64];      // [e][key] (transposed)
  __shared__ __align__(16) u16 lP[4 * 16 * 64];   // per-wave [m][key]

  const int tid  = threadIdx.x;
  const int lane = tid & 31, w = tid >> 5, hi = lane >> 4, nn = lane & 15;
  const size_t baseRow = (size_t)(blockIdx.y >> 4) * 2048;  // b*S
  const int col0 = (blockIdx.y & 15) * 64;                  // h*E
  const int q0   = blockIdx.x * 64;

  // stage Q once: straight copy -> async b128, 512 chunks, 4/thread
#pragma unroll
  for (int i = 0; i < 4; ++i) {
    int idx = tid + i * 128;                 // 0..511
    int r = idx >> 3, c = (idx & 7) << 3;    // 8 x 8-elem chunks per row
    async_b128(&lQ[r * 64 + c],
               qb + (baseRow + q0 + r) * 1024 + col0 + c);
  }
  wait_async0();
  __syncthreads();

  BFrag aq[2];  // Q A-fragments for E=64 (two K=32 steps), kept all kernel
  const int qrow = w * 16 + nn;
#pragma unroll
  for (int ke = 0; ke < 2; ++ke) {
    aq[ke].u[0] = *(const uint4*)&lQ[qrow * 64 + ke * 32 + hi * 8];
    aq[ke].u[1] = *(const uint4*)&lQ[qrow * 64 + ke * 32 + 16 + hi * 8];
  }

  v8f zero = {0.f, 0.f, 0.f, 0.f, 0.f, 0.f, 0.f, 0.f};
  v8f accO[4] = {zero, zero, zero, zero};
  float mrun[8], lrun[8];
#pragma unroll
  for (int r = 0; r < 8; ++r) { mrun[r] = -3.0e38f; lrun[r] = 0.f; }

  u16* pw = lP + w * (16 * 64);

  for (int kt = 0; kt < 32; ++kt) {
    __syncthreads();  // previous iter done with lK/lVT
    const int kbase = kt * 64;
    // K tile [key][e]: straight copy -> async b128, 512 chunks, 4/thread
#pragma unroll
    for (int i = 0; i < 4; ++i) {
      int idx = tid + i * 128;
      int r = idx >> 3, c = (idx & 7) << 3;
      async_b128(&lK[r * 64 + c],
                 kb + (baseRow + kbase + r) * 1024 + col0 + c);
    }
    // V tile transposed [e][key] (needs VGPR path)
#pragma unroll
    for (int i = 0; i < 32; ++i) {
      int idx = tid + i * 128;
      int r = idx >> 6, e = idx & 63;
      lVT[e * 64 + r] = vb[(baseRow + kbase + r) * 1024 + col0 + e];
    }
    wait_async0();
    __syncthreads();

    // S = Q * K^T : 4 tiles of 16x16, K-dim = E = 2 wmma steps
    v8f sc[4] = {zero, zero, zero, zero};
#pragma unroll
    for (int ke = 0; ke < 2; ++ke) {
#pragma unroll
      for (int nt = 0; nt < 4; ++nt) {
        BFrag bk;  // B[e][key]: lane n = key, K pairs contiguous in lK row
        const int kc = nt * 16 + nn;
        bk.u[0] = *(const uint4*)&lK[kc * 64 + ke * 32 + hi * 8];
        bk.u[1] = *(const uint4*)&lK[kc * 64 + ke * 32 + 16 + hi * 8];
        sc[nt] = wmma_bf16(aq[ke], bk, sc[nt]);
      }
    }

    // online softmax: rows live across 16-lane groups in C-layout
    float mnew[8], alpha[8];
#pragma unroll
    for (int r = 0; r < 8; ++r) {
      float mx = -3.0e38f;
#pragma unroll
      for (int nt = 0; nt < 4; ++nt) {
        sc[nt][r] *= 0.125f;  // 1/sqrt(E)
        mx = fmaxf(mx, sc[nt][r]);
      }
#pragma unroll
      for (int off = 1; off < 16; off <<= 1)
        mx = fmaxf(mx, __shfl_xor(mx, off));
      mnew[r]  = fmaxf(mrun[r], mx);
      alpha[r] = __expf(mrun[r] - mnew[r]);
      mrun[r]  = mnew[r];
    }

    float rs[8];
#pragma unroll
    for (int r = 0; r < 8; ++r) rs[r] = 0.f;
#pragma unroll
    for (int nt = 0; nt < 4; ++nt) {
#pragma unroll
      for (int r = 0; r < 8; ++r) {
        float p = __expf(sc[nt][r] - mnew[r]);
        rs[r] += p;
        pw[(r + hi * 8) * 64 + nt * 16 + nn] = f2bf(p);  // C->A via LDS
      }
    }
#pragma unroll
    for (int r = 0; r < 8; ++r) {
      float s = rs[r];
#pragma unroll
      for (int off = 1; off < 16; off <<= 1) s += __shfl_xor(s, off);
      lrun[r] = lrun[r] * alpha[r] + s;
#pragma unroll
      for (int et = 0; et < 4; ++et) accO[et][r] *= alpha[r];
    }

    // O += P * V : K-dim = 64 keys = 2 wmma steps
#pragma unroll
    for (int ke = 0; ke < 2; ++ke) {
      BFrag ap;
      ap.u[0] = *(const uint4*)&pw[nn * 64 + ke * 32 + hi * 8];
      ap.u[1] = *(const uint4*)&pw[nn * 64 + ke * 32 + 16 + hi * 8];
#pragma unroll
      for (int et = 0; et < 4; ++et) {
        BFrag bv;  // B[key][e] from lVT[e][key]: K pairs contiguous
        const int e = et * 16 + nn;
        bv.u[0] = *(const uint4*)&lVT[e * 64 + ke * 32 + hi * 8];
        bv.u[1] = *(const uint4*)&lVT[e * 64 + ke * 32 + 16 + hi * 8];
        accO[et] = wmma_bf16(ap, bv, accO[et]);
      }
    }
  }

  // finalize: O /= l, write bf16 [B*S][H*E]
#pragma unroll
  for (int r = 0; r < 8; ++r) {
    float  inv = 1.0f / lrun[r];
    size_t row = baseRow + q0 + w * 16 + r + hi * 8;
#pragma unroll
    for (int et = 0; et < 4; ++et)
      ob[row * 1024 + col0 + et * 16 + nn] = f2bf(accO[et][r] * inv);
  }
}

// -------------------------------------------------------------------- launch

extern "C" void kernel_launch(void* const* d_in, const int* in_sizes, int n_in,
                              void* d_out, int out_size, void* d_ws,
                              size_t ws_size, hipStream_t stream) {
  const float* x  = (const float*)d_in[0];
  const float* wq = (const float*)d_in[1];
  const float* wk = (const float*)d_in[2];
  const float* wv = (const float*)d_in[3];
  const float* wo = (const float*)d_in[4];
  float* out = (float*)d_out;

  // workspace layout (48 MB total)
  char* ws = (char*)d_ws;
  const size_t MB = 1ull << 20;
  u16* xb   = (u16*)(ws + 0 * MB);   // x bf16       [4096][1024]   8 MB
  u16* wqkv = (u16*)(ws + 8 * MB);   // fused qkv W  [1024][3072]   6 MB
  u16* wob  = (u16*)(ws + 14 * MB);  // wo bf16      [1024][1024]   2 MB
  u16* qb   = (u16*)(ws + 16 * MB);  // Q bf16       [4096][1024]   8 MB
  u16* kb   = (u16*)(ws + 24 * MB);  // K bf16                      8 MB
  u16* vb   = (u16*)(ws + 32 * MB);  // V bf16                      8 MB
  u16* ob   = (u16*)(ws + 40 * MB);  // O bf16                      8 MB

  k_f32_to_bf16<<<16384, 256, 0, stream>>>(x, xb, 4194304);
  k_pack_qkv<<<12288, 256, 0, stream>>>(wq, wk, wv, wqkv);
  k_f32_to_bf16<<<4096, 256, 0, stream>>>(wo, wob, 1048576);

  // QKV projection + RoPE: M=4096, N=3072, K=1024
  gemm_bf16_128x64<0><<<dim3(48, 32), 256, 0, stream>>>(
      xb, wqkv, 1024, 3072, qb, kb, vb, nullptr);

  // fused attention: grid (S/64, B*H)
  flash_attn<<<dim3(32, 32), 128, 0, stream>>>(qb, kb, vb, ob);

  // output projection: M=4096, N=1024, K=1024 -> fp32 d_out
  gemm_bf16_128x64<1><<<dim3(16, 32), 256, 0, stream>>>(
      ob, wob, 1024, 1024, nullptr, nullptr, nullptr, out);
}
